// GCN1_1872605741060
// MI455X (gfx1250) — matrix operand order repaired and checked by reading, not compiled, not run on hardware
//
#include <hip/hip_runtime.h>
#include <cstdint>
#include <cstddef>

typedef __attribute__((ext_vector_type(16))) __bf16       v16bf;
typedef __attribute__((ext_vector_type(8)))  float        v8f;
typedef __attribute__((ext_vector_type(4)))  unsigned int u32x4;

union Frag  { v16bf v; u32x4 q[2]; };
union Pack8 { __bf16 h[8]; u32x4 q; };

#define MREAL 10000
#define MP    10112   /* 79 * 128, multiple of 128 (M tiles) and 32 (K step) */

__device__ __forceinline__ unsigned hash_u32(unsigned x) {
    x *= 0x9E3779B1u; x ^= x >> 16;
    x *= 0x85EBCA6Bu; x ^= x >> 13;
    x *= 0xC2B2AE35u; x ^= x >> 16;
    return x;
}

// C[MP x N] = A[MP x Kp] (row-major bf16) @ BT[N x Kp]^T (K-major bf16)
// MODE 0: store Cb transposed (N-major, ld = MP), no bias   -> next GEMM's BT
// MODE 1: store Cb row-major, bias + relu + dropout(0.5)
// MODE 2: store Cb row-major, bias
// MODE 3: store Cf row-major fp32, bias, row guard (< MREAL)
template<int MODE>
__global__ __launch_bounds__(256)
void wmma_gemm(const __bf16* __restrict__ A, const __bf16* __restrict__ BT,
               const float* __restrict__ bias, float* __restrict__ Cf,
               __bf16* __restrict__ Cb, int N, int Kp)
{
    __shared__ __align__(16) __bf16 a_lds[2][128 * 40];
    __shared__ __align__(16) __bf16 b_lds[2][128 * 40];

    const int tid    = threadIdx.x;
    const int lane   = tid & 31;
    const int wave   = tid >> 5;
    const int wm     = wave & 3;     // 0..3 -> 32-row slab of M
    const int wn     = wave >> 2;    // 0..1 -> 64-col slab of N
    const int lane16 = lane & 15;
    const int half   = lane >> 4;

    const int bn0 = blockIdx.x * 128;
    const int bm0 = blockIdx.y * 128;

    // global->LDS staging: each thread moves 2x16B per tile per matrix
    const int lrow0 = tid >> 2;          // 0..63 (rows 0-63, +64 for second pass)
    const int lcol  = (tid & 3) * 8;     // bf16 element offset: 0,8,16,24

    const __bf16* gA = A  + (size_t)(bm0 + lrow0) * Kp + lcol;
    const __bf16* gB = BT + (size_t)(bn0 + lrow0) * Kp + lcol;
    const size_t rowskip = (size_t)64 * Kp;

    v8f acc[2][4];
#pragma unroll
    for (int mi = 0; mi < 2; ++mi)
#pragma unroll
        for (int ni = 0; ni < 4; ++ni)
#pragma unroll
            for (int r = 0; r < 8; ++r) acc[mi][ni][r] = 0.0f;

    // ---- prologue: stage k-tile 0 ----
    u32x4 ra0 = *(const u32x4*)(gA);
    u32x4 ra1 = *(const u32x4*)(gA + rowskip);
    u32x4 rb0 = *(const u32x4*)(gB);
    u32x4 rb1 = *(const u32x4*)(gB + rowskip);
    *(u32x4*)&a_lds[0][lrow0 * 40 + lcol]        = ra0;
    *(u32x4*)&a_lds[0][(lrow0 + 64) * 40 + lcol] = ra1;
    *(u32x4*)&b_lds[0][lrow0 * 40 + lcol]        = rb0;
    *(u32x4*)&b_lds[0][(lrow0 + 64) * 40 + lcol] = rb1;
    __syncthreads();

    const int KT = Kp >> 5;
    for (int kt = 0; kt < KT; ++kt) {
        const int cur = kt & 1;

        // issue global loads for next k-tile early (overlap with WMMAs)
        if (kt + 1 < KT) {
            const __bf16* pA = gA + (size_t)(kt + 1) * 32;
            const __bf16* pB = gB + (size_t)(kt + 1) * 32;
            ra0 = *(const u32x4*)(pA);
            ra1 = *(const u32x4*)(pA + rowskip);
            rb0 = *(const u32x4*)(pB);
            rb1 = *(const u32x4*)(pB + rowskip);
        }

        // fragment loads per ISA 7.12.2 bf16 layouts
        Frag fa[2], fb[4];
        {
            const __bf16* base = &a_lds[cur][(wm * 32 + lane16) * 40 + half * 8];
            fa[0].q[0] = *(const u32x4*)(base);
            fa[0].q[1] = *(const u32x4*)(base + 16);
            fa[1].q[0] = *(const u32x4*)(base + 16 * 40);
            fa[1].q[1] = *(const u32x4*)(base + 16 * 40 + 16);
        }
#pragma unroll
        for (int ni = 0; ni < 4; ++ni) {
            const __bf16* base = &b_lds[cur][(wn * 64 + ni * 16 + lane16) * 40 + half * 8];
            fb[ni].q[0] = *(const u32x4*)(base);
            fb[ni].q[1] = *(const u32x4*)(base + 16);
        }

#pragma unroll
        for (int mi = 0; mi < 2; ++mi)
#pragma unroll
            for (int ni = 0; ni < 4; ++ni)
                acc[mi][ni] = __builtin_amdgcn_wmma_f32_16x16x32_bf16(
                    false, fa[mi].v, false, fb[ni].v,
                    (short)0, acc[mi][ni], false, false);

        __syncthreads();
        if (kt + 1 < KT) {
            const int nxt = cur ^ 1;
            *(u32x4*)&a_lds[nxt][lrow0 * 40 + lcol]        = ra0;
            *(u32x4*)&a_lds[nxt][(lrow0 + 64) * 40 + lcol] = ra1;
            *(u32x4*)&b_lds[nxt][lrow0 * 40 + lcol]        = rb0;
            *(u32x4*)&b_lds[nxt][(lrow0 + 64) * 40 + lcol] = rb1;
        }
        __syncthreads();
    }

    // ---- epilogue ----
    const int m_base = bm0 + wm * 32 + half * 8;  // + mi*16 + r
    const int n_base = bn0 + wn * 64 + lane16;    // + ni*16

    if (MODE == 0) {
#pragma unroll
        for (int mi = 0; mi < 2; ++mi)
#pragma unroll
            for (int ni = 0; ni < 4; ++ni) {
                Pack8 p;
#pragma unroll
                for (int r = 0; r < 8; ++r) p.h[r] = (__bf16)acc[mi][ni][r];
                const int gn = n_base + ni * 16;
                const int gm = m_base + mi * 16;
                *(u32x4*)&Cb[(size_t)gn * MP + gm] = p.q;
            }
    } else {
#pragma unroll
        for (int ni = 0; ni < 4; ++ni) {
            const int gn = n_base + ni * 16;
            const float bv = bias[gn];
#pragma unroll
            for (int mi = 0; mi < 2; ++mi) {
#pragma unroll
                for (int r = 0; r < 8; ++r) {
                    const int gm = m_base + mi * 16 + r;
                    float f = acc[mi][ni][r] + bv;
                    if (MODE == 1) {
                        f = f > 0.0f ? f : 0.0f;
                        const unsigned h = hash_u32((unsigned)gm * (unsigned)N + (unsigned)gn);
                        f = (h & 1u) ? f * 2.0f : 0.0f;
                    }
                    if (MODE == 3) {
                        if (gm < MREAL) Cf[(size_t)gm * N + gn] = f;
                    } else {
                        Cb[(size_t)gm * N + gn] = (__bf16)f;
                    }
                }
            }
        }
    }
}

// fp32 -> bf16 with zero padding to (dstR x dstC)
__global__ __launch_bounds__(256)
void cvt_pad_bf16(const float* __restrict__ src, __bf16* __restrict__ dst,
                  int srcR, int srcC, int dstR, int dstC)
{
    const int cols8 = dstC >> 3;
    const int idx = blockIdx.x * blockDim.x + threadIdx.x;
    if (idx >= dstR * cols8) return;
    const int r  = idx / cols8;
    const int c0 = (idx - r * cols8) << 3;
    Pack8 p;
    if (r < srcR && c0 + 8 <= srcC) {
        const float* s = src + (size_t)r * srcC + c0;
#pragma unroll
        for (int i = 0; i < 8; ++i) p.h[i] = (__bf16)s[i];
    } else {
#pragma unroll
        for (int i = 0; i < 8; ++i) {
            const float v = (r < srcR && (c0 + i) < srcC)
                                ? src[(size_t)r * srcC + c0 + i] : 0.0f;
            p.h[i] = (__bf16)v;
        }
    }
    *(u32x4*)&dst[(size_t)r * dstC + c0] = p.q;
}

// W (K x N fp32, row-major) -> WT (N x K bf16, K-major)
__global__ __launch_bounds__(256)
void cvt_transpose_bf16(const float* __restrict__ src, __bf16* __restrict__ dst,
                        int K, int N)
{
    const int idx = blockIdx.x * blockDim.x + threadIdx.x;
    if (idx >= K * N) return;
    const int n = idx / K;
    const int k = idx - n * K;
    dst[idx] = (__bf16)src[(size_t)k * N + n];
}

extern "C" void kernel_launch(void* const* d_in, const int* in_sizes, int n_in,
                              void* d_out, int out_size, void* d_ws, size_t ws_size,
                              hipStream_t stream)
{
    (void)in_sizes; (void)n_in; (void)out_size; (void)ws_size;
    const float* x   = (const float*)d_in[0];
    const float* adj = (const float*)d_in[1];
    const float* W1  = (const float*)d_in[2];
    const float* b1  = (const float*)d_in[3];
    const float* W2  = (const float*)d_in[4];
    const float* b2  = (const float*)d_in[5];
    const float* W3  = (const float*)d_in[6];
    const float* b3  = (const float*)d_in[7];
    float* out = (float*)d_out;

    char* ws = (char*)d_ws;
    size_t off = 0;
    auto take = [&](size_t bytes) -> char* {
        char* p = ws + off;
        off = (off + bytes + 255) & ~(size_t)255;
        return p;
    };
    __bf16* adjb = (__bf16*)take((size_t)MP * MP * 2);
    __bf16* xb   = (__bf16*)take((size_t)MP * 1024 * 2);
    __bf16* w1t  = (__bf16*)take((size_t)512 * 1024 * 2);
    __bf16* w2t  = (__bf16*)take((size_t)256 * 512 * 2);
    __bf16* w3t  = (__bf16*)take((size_t)128 * 256 * 2);
    __bf16* h1t  = (__bf16*)take((size_t)512 * MP * 2);
    __bf16* x1   = (__bf16*)take((size_t)MP * 512 * 2);
    __bf16* h2t  = (__bf16*)take((size_t)256 * MP * 2);
    __bf16* x2   = (__bf16*)take((size_t)MP * 256 * 2);
    __bf16* h3t  = (__bf16*)take((size_t)128 * MP * 2);

    // ---- precision conversion (adj dominates: 400MB read, 200MB write) ----
    {
        int total = MP * (MP >> 3);
        cvt_pad_bf16<<<(total + 255) / 256, 256, 0, stream>>>(adj, adjb, MREAL, MREAL, MP, MP);
        total = MP * (1024 >> 3);
        cvt_pad_bf16<<<(total + 255) / 256, 256, 0, stream>>>(x, xb, MREAL, 1024, MP, 1024);
        cvt_transpose_bf16<<<(1024 * 512 + 255) / 256, 256, 0, stream>>>(W1, w1t, 1024, 512);
        cvt_transpose_bf16<<<(512 * 256 + 255) / 256, 256, 0, stream>>>(W2, w2t, 512, 256);
        cvt_transpose_bf16<<<(256 * 128 + 255) / 256, 256, 0, stream>>>(W3, w3t, 256, 128);
    }

    const dim3 blk(256);
    const int mt = MP / 128;  // 79
    // layer 1: H1 = x @ W1                 -> transposed bf16 (B operand of next)
    wmma_gemm<0><<<dim3(512 / 128, mt), blk, 0, stream>>>(xb,   w1t, nullptr, nullptr, h1t, 512, 1024);
    // X1 = dropout(relu(adj @ H1 + b1))    -> row-major bf16 (A operand of next)
    wmma_gemm<1><<<dim3(512 / 128, mt), blk, 0, stream>>>(adjb, h1t, b1,      nullptr, x1,  512, MP);
    // layer 2: H2 = X1 @ W2
    wmma_gemm<0><<<dim3(256 / 128, mt), blk, 0, stream>>>(x1,   w2t, nullptr, nullptr, h2t, 256, 512);
    // X2 = adj @ H2 + b2
    wmma_gemm<2><<<dim3(256 / 128, mt), blk, 0, stream>>>(adjb, h2t, b2,      nullptr, x2,  256, MP);
    // layer 3: H3 = X2 @ W3
    wmma_gemm<0><<<dim3(128 / 128, mt), blk, 0, stream>>>(x2,   w3t, nullptr, nullptr, h3t, 128, 256);
    // out = adj @ H3 + b3 (fp32, row guard)
    wmma_gemm<3><<<dim3(128 / 128, mt), blk, 0, stream>>>(adjb, h3t, b3,      out,     nullptr, 128, MP);
}
